// IBEBM_60266981097570
// MI455X (gfx1250) — matrix operand order repaired and checked by reading, not compiled, not run on hardware
//
#include <hip/hip_runtime.h>
#include <hip/hip_bf16.h>
#include <math.h>

// ---------------------------------------------------------------------------
// IB-EBM VAE training-loss pipeline for MI455X (gfx1250).
//
// Roofline: ~650 GFLOP of bf16-WMMA GEMM; weights (~12MB bf16) and activations
// (64MB bf16 each) are L2-resident (192MB), so the fight is L2 bandwidth vs
// matrix math. Big GEMMs use 128x128 LDS block tiles, double-buffered via
// GLOBAL_LOAD_ASYNC_TO_LDS_B128 (ASYNCcnt / s_wait_asynccnt), 8 waves x
// (32x64) register tiles -> 8 v_wmma_f32_16x16x32_bf16 per wave per K-slab,
// ~16KB of global traffic per 64-WMMA slab (vs 96KB/2-WMMA direct).
// Small-N GEMMs (N=32/20) keep the direct wave-tile path.
// ---------------------------------------------------------------------------

typedef unsigned short u16;
typedef unsigned int   u32;
typedef unsigned long long u64;
typedef __attribute__((ext_vector_type(4)))  u32    v4u;
typedef __attribute__((ext_vector_type(16))) __bf16 v16bf;
typedef __attribute__((ext_vector_type(8)))  float  v8f;

#define B_N    32768
#define XD_N   1024
#define ZD_N   32
#define NC_N   20
#define NSTEPS 20

enum { ACC_SQ = 0, ACC_KL = 1, ACC_POS = 2, ACC_NEG = 3, ACC_HYZ = 4,
       ACC_MEANP = 8, ACC_COUNT = 64 };

enum { EPI_RELU_BF16 = 0,   // out0.bf16 = relu(acc+bias)
       EPI_LIN_BF16,        // out0.bf16 = acc+bias
       EPI_LIN_F32,         // out0.f32  = acc+bias         (bias may be null)
       EPI_GELU_PREACT,     // out1.bf16 = acc+bias ; out0.bf16 = gelu(acc+bias)
       EPI_SQDIFF,          // accum += sum((aux.f32 - (acc+bias))^2)  (no store)
       EPI_GELUGRAD };      // out0.bf16 = acc * gelu'(aux.bf16)

// ---------------- scalar helpers ----------------
__device__ __forceinline__ float bf2f(u16 v) { return __uint_as_float(((u32)v) << 16); }
__device__ __forceinline__ u16 f2bf(float f) {
  u32 u = __float_as_uint(f);
  u32 r = u + 0x7fffu + ((u >> 16) & 1u);   // round-to-nearest-even
  return (u16)(r >> 16);
}
__device__ __forceinline__ float geluf(float x) {
  return 0.5f * x * (1.0f + erff(x * 0.70710678118f));
}
__device__ __forceinline__ float gelugradf(float x) {
  return 0.5f * (1.0f + erff(x * 0.70710678118f)) +
         x * 0.3989422804f * __expf(-0.5f * x * x);
}
__device__ __forceinline__ u32 hash32(u32 x) {
  x ^= x >> 16; x *= 0x7feb352du; x ^= x >> 15; x *= 0x846ca68bu; x ^= x >> 16;
  return x;
}
// Deterministic standard normal (hash + Box-Muller). Same value every call.
__device__ __forceinline__ float rnorm(u32 idx, u32 salt) {
  u32 a = hash32(idx * 2u + 1u + salt * 0x9e3779b9u);
  u32 b = hash32(idx * 2u + 2u + salt * 0x85ebca6bu);
  float u1 = (float)((a >> 8) + 1u) * (1.0f / 16777217.0f);   // (0,1]
  float u2 = (float)(b >> 8) * (1.0f / 16777216.0f);          // [0,1)
  return sqrtf(-2.0f * __logf(u1)) * __cosf(6.28318530718f * u2);
}

// 256-thread block sum; result valid on thread 0. Safe to call twice (syncs).
__device__ __forceinline__ float block_sum256(float v) {
  __shared__ float red[8];
  #pragma unroll
  for (int o = 16; o > 0; o >>= 1) v += __shfl_down(v, o);
  __syncthreads();
  if ((threadIdx.x & 31u) == 0) red[threadIdx.x >> 5] = v;
  __syncthreads();
  float r = 0.f;
  if (threadIdx.x == 0) {
    #pragma unroll
    for (int i = 0; i < 8; i++) r += red[i];
  }
  return r;
}

// gfx1250 async global->LDS copy, 16B per lane. ldsOff = LDS byte address
// (addr[31:0] of a __shared__ pointer), ga = 64-bit global address (GV mode).
__device__ __forceinline__ void async_g2l_b128(u32 ldsOff, u64 ga) {
  asm volatile("global_load_async_to_lds_b128 %0, %1, off"
               :: "v"(ldsOff), "v"(ga) : "memory");
}
__device__ __forceinline__ void wait_async0() {
  asm volatile("s_wait_asynccnt 0" ::: "memory");
}

// ---------------------------------------------------------------------------
// LDS-tiled WMMA GEMM (N, M multiples of 128; K multiple of 32).
//   out[M, ldc] = epilogue(A[M,K](bf16) @ Wt[N,K](bf16, = W^T) + bias)
// Block: 256 threads = 8 waves; 128x128 block tile; wave grid 4(M) x 2(N);
// each wave owns a 32x64 register tile = 2 A-frags x 4 B-frags -> 8 WMMA/slab.
// A/B K-slabs (128x32) are staged to LDS with async b128 copies, double-
// buffered; fragments are read from LDS per ISA 7.12.2 wave32 layouts:
//   A 16x32: lane r=lane&15 -> row; h=lane>>4: elems 0..7 <- K=h*8+0..7,
//            elems 8..15 <- K=16+h*8+0..7
//   B 32x16: lane -> col (r), elems 0..7 <- K=h*8+0..7, 8..15 <- K=16+h*8+0..7
//   C/D f32: lane r -> col, VGPR j -> row h*8+j
// ---------------------------------------------------------------------------
#define LDSROW 40   // 32 halves + 8 pad (80B row stride: spreads LDS banks)

template <int EPI>
__global__ __launch_bounds__(256)
void gemm_wmma_lds(const u16* __restrict__ A, const u16* __restrict__ Wt,
                   const float* __restrict__ bias,
                   void* __restrict__ out0, void* __restrict__ out1,
                   const void* __restrict__ aux, float* __restrict__ accum,
                   int M, int N, int K, int ldc) {
  __shared__ u16 As[2][128 * LDSROW];
  __shared__ u16 Bs[2][128 * LDSROW];

  const int t    = (int)threadIdx.x;
  const int lane = t & 31, wid = t >> 5;
  const int wr = wid & 3, wc = wid >> 2;   // wave position: 4 x 2
  const int nb = N >> 7;
  const int bm = (int)blockIdx.x / nb, bn = (int)blockIdx.x % nb;
  const int m0 = bm << 7, n0 = bn << 7;
  const int r = lane & 15, h = lane >> 4;

  // Staging: 512 x 16B chunks per 128x32 slab; thread t copies chunks t, t+256.
  const int row0 = t >> 2,        prt0 = t & 3;
  const int row1 = (t + 256) >> 2, prt1 = t & 3;   // (t+256)&3 == t&3
  const long gA0 = (long)(m0 + row0) * K + prt0 * 8;
  const long gA1 = (long)(m0 + row1) * K + prt1 * 8;
  const long gB0 = (long)(n0 + row0) * K + prt0 * 8;
  const long gB1 = (long)(n0 + row1) * K + prt1 * 8;
  const int lOff0 = row0 * LDSROW + prt0 * 8;      // in halves
  const int lOff1 = row1 * LDSROW + prt1 * 8;

  auto stage = [&](int buf, int k0) {
    async_g2l_b128((u32)(size_t)&As[buf][lOff0], (u64)(A  + gA0 + k0));
    async_g2l_b128((u32)(size_t)&As[buf][lOff1], (u64)(A  + gA1 + k0));
    async_g2l_b128((u32)(size_t)&Bs[buf][lOff0], (u64)(Wt + gB0 + k0));
    async_g2l_b128((u32)(size_t)&Bs[buf][lOff1], (u64)(Wt + gB1 + k0));
  };

  stage(0, 0);
  wait_async0();
  __syncthreads();

  v8f acc8[2][4] = {};
  int buf = 0;
  for (int k0 = 0; k0 < K; k0 += 32) {
    if (k0 + 32 < K) stage(buf ^ 1, k0 + 32);     // prefetch next slab

    v16bf af[2], bfr[4];
    #pragma unroll
    for (int m = 0; m < 2; m++) {
      const u16* p = &As[buf][(wr * 32 + m * 16 + r) * LDSROW + h * 8];
      ((v4u*)&af[m])[0] = *(const v4u*)(p);
      ((v4u*)&af[m])[1] = *(const v4u*)(p + 16);
    }
    #pragma unroll
    for (int n = 0; n < 4; n++) {
      const u16* p = &Bs[buf][(wc * 64 + n * 16 + r) * LDSROW + h * 8];
      ((v4u*)&bfr[n])[0] = *(const v4u*)(p);
      ((v4u*)&bfr[n])[1] = *(const v4u*)(p + 16);
    }
    #pragma unroll
    for (int m = 0; m < 2; m++) {
      #pragma unroll
      for (int n = 0; n < 4; n++) {
        acc8[m][n] = __builtin_amdgcn_wmma_f32_16x16x32_bf16(
            false, af[m], false, bfr[n], (short)0, acc8[m][n], false, false);
      }
    }
    wait_async0();      // our async copies into buf^1 have landed
    __syncthreads();    // everyone done reading buf & done staging buf^1
    buf ^= 1;
  }

  // ---- epilogue ----
  float sq = 0.f;
  #pragma unroll
  for (int n = 0; n < 4; n++) {
    const int col = n0 + wc * 64 + n * 16 + r;
    const float bv = bias ? bias[col] : 0.f;
    #pragma unroll
    for (int m = 0; m < 2; m++) {
      const int rowB = m0 + wr * 32 + m * 16 + h * 8;
      #pragma unroll
      for (int j = 0; j < 8; j++) {
        const float v = acc8[m][n][j] + bv;
        const long idx = (long)(rowB + j) * ldc + col;
        if (EPI == EPI_RELU_BF16) {
          ((u16*)out0)[idx] = f2bf(fmaxf(v, 0.f));
        } else if (EPI == EPI_LIN_BF16) {
          ((u16*)out0)[idx] = f2bf(v);
        } else if (EPI == EPI_GELU_PREACT) {
          ((u16*)out1)[idx] = f2bf(v);
          ((u16*)out0)[idx] = f2bf(geluf(v));
        } else if (EPI == EPI_SQDIFF) {
          const float d = ((const float*)aux)[idx] - v;
          sq += d * d;
        } else if (EPI == EPI_GELUGRAD) {
          const float p = bf2f(((const u16*)aux)[idx]);
          ((u16*)out0)[idx] = f2bf(v * gelugradf(p));
        }
      }
    }
  }
  if (EPI == EPI_SQDIFF) {
    #pragma unroll
    for (int o = 16; o > 0; o >>= 1) sq += __shfl_down(sq, o);
    if (lane == 0) atomicAdd(accum, sq);
  }
}

// ---------------------------------------------------------------------------
// Direct wave-tile WMMA GEMM for small-N panels (N=32 padded; B is L2-hot).
// One wave = 16x32 output tile, fragments straight from global.
// ---------------------------------------------------------------------------
template <int EPI>
__global__ __launch_bounds__(256)
void gemm_wmma(const u16* __restrict__ A, const u16* __restrict__ Wt,
               const float* __restrict__ bias,
               void* __restrict__ out0, void* __restrict__ out1,
               const void* __restrict__ aux, float* __restrict__ accum,
               int M, int Npad, int K, int Nstore, int ldc) {
  const int lane = (int)(threadIdx.x & 31u);
  const int wid  = (int)(threadIdx.x >> 5);
  const int tilesN = Npad >> 5;
  const long tid   = (long)blockIdx.x * 8 + wid;
  const long total = (long)(M >> 4) * tilesN;
  if (tid >= total) return;              // uniform per wave: EXEC stays all-1s
  const int tn = (int)(tid % tilesN), tm = (int)(tid / tilesN);
  const int m0 = tm << 4, n0 = tn << 5;
  const int r = lane & 15, h = lane >> 4;

  const u16* pA  = A  + (long)(m0 + r) * K + h * 8;
  const u16* pB0 = Wt + (long)(n0 + r) * K + h * 8;
  const u16* pB1 = pB0 + (long)16 * K;

  v8f c0 = {}; v8f c1 = {};
  for (int k = 0; k < K; k += 32) {
    v16bf a, b0, b1;
    ((v4u*)&a)[0]  = *(const v4u*)(pA  + k);
    ((v4u*)&a)[1]  = *(const v4u*)(pA  + k + 16);
    ((v4u*)&b0)[0] = *(const v4u*)(pB0 + k);
    ((v4u*)&b0)[1] = *(const v4u*)(pB0 + k + 16);
    ((v4u*)&b1)[0] = *(const v4u*)(pB1 + k);
    ((v4u*)&b1)[1] = *(const v4u*)(pB1 + k + 16);
    if (k + 64 <= K) {
      __builtin_prefetch(pA  + k + 32, 0, 0);
      __builtin_prefetch(pB0 + k + 32, 0, 0);
      __builtin_prefetch(pB1 + k + 32, 0, 0);
    }
    c0 = __builtin_amdgcn_wmma_f32_16x16x32_bf16(false, a, false, b0,
                                                 (short)0, c0, false, false);
    c1 = __builtin_amdgcn_wmma_f32_16x16x32_bf16(false, a, false, b1,
                                                 (short)0, c1, false, false);
  }

  const int col0 = n0 + r;
  const int col1 = col0 + 16;
  const int rowBase = m0 + h * 8;
  float b0v = 0.f, b1v = 0.f;
  if (bias) {
    if (col0 < Nstore) b0v = bias[col0];
    if (col1 < Nstore) b1v = bias[col1];
  }
  #pragma unroll
  for (int j = 0; j < 8; j++) {
    const int row = rowBase + j;
    const float v0 = c0[j] + b0v;
    const float v1 = c1[j] + b1v;
    const long i0 = (long)row * ldc + col0;
    const long i1 = (long)row * ldc + col1;
    if (EPI == EPI_LIN_F32) {
      if (col0 < Nstore) ((float*)out0)[i0] = v0;
      if (col1 < Nstore) ((float*)out0)[i1] = v1;
    } else if (EPI == EPI_LIN_BF16) {
      if (col0 < Nstore) ((u16*)out0)[i0] = f2bf(v0);
      if (col1 < Nstore) ((u16*)out0)[i1] = f2bf(v1);
    }
  }
  (void)out1; (void)aux; (void)accum;
}

// ---------------- elementwise / prep kernels ----------------
__global__ void k_zero(float* p, int n) {
  int i = blockIdx.x * blockDim.x + threadIdx.x;
  if (i < n) p[i] = 0.f;
}

__global__ void k_cvt_bf16(const float* __restrict__ s, u16* __restrict__ d, long n) {
  long i = (long)blockIdx.x * 256 + threadIdx.x;
  if (i < n) d[i] = f2bf(s[i]);
}

// t[n*Kpad + k] = (k<K && n<N) ? w[k*N + n] : 0      (w is K x N row-major)
__global__ void k_transpose_bf16(const float* __restrict__ w, u16* __restrict__ t,
                                 int K, int N, int Kpad, int Npad) {
  long i = (long)blockIdx.x * 256 + threadIdx.x;
  long tot = (long)Kpad * Npad;
  if (i >= tot) return;
  int n = (int)(i / Kpad), k = (int)(i % Kpad);
  float v = (k < K && n < N) ? w[(long)k * N + n] : 0.f;
  t[i] = f2bf(v);
}

// t[n*Kpad + k] = (k<K) ? w[n*K + k] : 0             (w is rows x K row-major)
__global__ void k_padcopy_bf16(const float* __restrict__ w, u16* __restrict__ t,
                               int K, int Kpad, long rows) {
  long i = (long)blockIdx.x * 256 + threadIdx.x;
  long tot = rows * Kpad;
  if (i >= tot) return;
  int n = (int)(i / Kpad), k = (int)(i % Kpad);
  t[i] = f2bf((k < K) ? w[(long)n * K + k] : 0.f);
}

// z = mu + exp(0.5*lv)*eps ; accumulate KL.   Grid covers exactly B*ZD.
__global__ __launch_bounds__(256)
void k_reparam_kl(const float* __restrict__ mu, const float* __restrict__ lv,
                  u16* __restrict__ zb, float* __restrict__ acc) {
  int i = blockIdx.x * 256 + threadIdx.x;
  const float m = mu[i], l = lv[i];
  const float z = m + __expf(0.5f * l) * rnorm((u32)i, 1u);
  zb[i] = f2bf(z);
  const float kle = -0.5f * (1.0f + l - m * m - __expf(l));
  const float s = block_sum256(kle);
  if (threadIdx.x == 0) atomicAdd(&acc[ACC_KL], s);
}

__global__ void k_init_z(float* __restrict__ zpr, u16* __restrict__ zb) {
  int i = blockIdx.x * 256 + threadIdx.x;
  const float z = rnorm((u32)i, 2u);
  zpr[i] = z;
  zb[i] = f2bf(z);
}

// z <- z - 0.5*STEP^2*(g + z) + STEP*noise   (STEP=0.4, SIG=1)
__global__ void k_langevin(float* __restrict__ zpr, const float* __restrict__ dz,
                           u16* __restrict__ zb, u32 salt) {
  int i = blockIdx.x * 256 + threadIdx.x;
  float z = zpr[i];
  const float g = dz[i];
  z = z - 0.08f * (g + z) + 0.4f * rnorm((u32)i, salt);
  zpr[i] = z;
  zb[i] = f2bf(z);
}

// Per-row logsumexp / softmax over NC=20 logits.
//   lse_acc : accumulate sum of lse               (prob_pos / prob_neg)
//   dlog    : write -softmax (bf16, padded to 32) (MCMC backward)
//   acc_mi  : accumulate column sums of p (mean_p) and -sum(p*logp) (H_y_z)
__global__ __launch_bounds__(256)
void k_lse(const float* __restrict__ logits, float* __restrict__ lse_acc,
           u16* __restrict__ dlog, float* __restrict__ acc_mi) {
  int i = blockIdx.x * 256 + threadIdx.x;  // row
  float l[NC_N];
  float m = -1e30f;
  #pragma unroll
  for (int c = 0; c < NC_N; c++) { l[c] = logits[(long)i * NC_N + c]; m = fmaxf(m, l[c]); }
  float se = 0.f;
  #pragma unroll
  for (int c = 0; c < NC_N; c++) { l[c] = __expf(l[c] - m); se += l[c]; }
  const float lse = m + __logf(se);
  const float inv = 1.0f / se;

  __shared__ float colp[NC_N];
  if (acc_mi) {
    if (threadIdx.x < NC_N) colp[threadIdx.x] = 0.f;
    __syncthreads();
  }
  float hyz = 0.f;
  #pragma unroll
  for (int c = 0; c < NC_N; c++) {
    const float p = l[c] * inv;
    if (dlog) dlog[(long)i * 32 + c] = f2bf(-p);
    if (acc_mi) {
      const float lp = __logf(fmaxf(p, 1e-30f));
      hyz += -p * lp;
      atomicAdd(&colp[c], p);
    }
  }
  if (dlog) {
    #pragma unroll
    for (int c = NC_N; c < 32; c++) dlog[(long)i * 32 + c] = 0;
  }
  if (acc_mi) {
    __syncthreads();
    if (threadIdx.x < NC_N) atomicAdd(&acc_mi[ACC_MEANP + threadIdx.x], colp[threadIdx.x]);
    const float s = block_sum256(hyz);
    if (threadIdx.x == 0) atomicAdd(&acc_mi[ACC_HYZ], s);
  }
  if (lse_acc) {
    const float s = block_sum256(lse);
    if (threadIdx.x == 0) atomicAdd(lse_acc, s);
  }
}

__global__ void k_finalize(const float* __restrict__ acc, float* __restrict__ out) {
  const float fB = (float)B_N;
  const float recon = acc[ACC_SQ] / (fB * (float)XD_N);
  const float kl    = acc[ACC_KL]  / fB;
  const float pos   = acc[ACC_POS] / fB;
  const float neg   = acc[ACC_NEG] / fB;
  const float hyz   = acc[ACC_HYZ] / fB;
  float hy = 0.f;
  for (int c = 0; c < NC_N; c++) {
    const float py = acc[ACC_MEANP + c] / fB;
    const float lp = __logf(py + 1e-15f);
    hy += -__expf(lp) * lp;
  }
  const float mi = hy - hyz;
  out[0] = recon + kl - pos + neg - 15.0f * mi;
}

// ---------------- host-side helpers ----------------
template <int EPI>
static inline void run_lds(const u16* A, const u16* Wt, const float* bias,
                           void* o0, void* o1, const void* aux, float* acc,
                           int M, int N, int K, int ldc, hipStream_t st) {
  const int blocks = (M / 128) * (N / 128);
  gemm_wmma_lds<EPI><<<blocks, 256, 0, st>>>(A, Wt, bias, o0, o1, aux, acc,
                                             M, N, K, ldc);
}

template <int EPI>
static inline void run_direct(const u16* A, const u16* Wt, const float* bias,
                              void* o0, int M, int Npad, int K, int Nstore,
                              int ldc, hipStream_t st) {
  const long tiles = (long)(M / 16) * (Npad / 32);
  const int blocks = (int)((tiles + 7) / 8);
  gemm_wmma<EPI><<<blocks, 256, 0, st>>>(A, Wt, bias, o0, nullptr, nullptr,
                                         nullptr, M, Npad, K, Nstore, ldc);
}

extern "C" void kernel_launch(void* const* d_in, const int* in_sizes, int n_in,
                              void* d_out, int out_size, void* d_ws, size_t ws_size,
                              hipStream_t stream) {
  (void)in_sizes; (void)n_in; (void)out_size; (void)ws_size;

  const float* x      = (const float*)d_in[0];
  const float* enc1_w = (const float*)d_in[1];  const float* enc1_b = (const float*)d_in[2];
  const float* enc2_w = (const float*)d_in[3];  const float* enc2_b = (const float*)d_in[4];
  const float* enc3_w = (const float*)d_in[5];  const float* enc3_b = (const float*)d_in[6];
  const float* mu_w   = (const float*)d_in[7];  const float* mu_b   = (const float*)d_in[8];
  const float* lv_w   = (const float*)d_in[9];  const float* lv_b   = (const float*)d_in[10];
  const float* dec1_w = (const float*)d_in[11]; const float* dec1_b = (const float*)d_in[12];
  const float* dec2_w = (const float*)d_in[13]; const float* dec2_b = (const float*)d_in[14];
  const float* dec3_w = (const float*)d_in[15]; const float* dec3_b = (const float*)d_in[16];
  const float* dec4_w = (const float*)d_in[17]; const float* dec4_b = (const float*)d_in[18];
  const float* ebm1_w = (const float*)d_in[19]; const float* ebm1_b = (const float*)d_in[20];
  const float* ebm2_w = (const float*)d_in[21]; const float* ebm2_b = (const float*)d_in[22];

  // ---- workspace carve-up ----
  char* base = (char*)d_ws;
  size_t off = 0;
  auto take = [&](size_t bytes) -> void* {
    void* p = base + off;
    off = (off + bytes + 255) & ~(size_t)255;
    return p;
  };
  u16* bufA = (u16*)take((size_t)B_N * 1024 * 2);   // rotating activation bufs
  u16* bufB = (u16*)take((size_t)B_N * 1024 * 2);
  u16* bufC = (u16*)take((size_t)B_N * 1024 * 2);
  u16* WtE1 = (u16*)take((size_t)1024 * 1024 * 2);  // bf16, N-major (W^T)
  u16* WtE2 = (u16*)take((size_t)1024 * 1024 * 2);
  u16* WtE3 = (u16*)take((size_t)1024 * 1024 * 2);
  u16* WtD2 = (u16*)take((size_t)1024 * 1024 * 2);
  u16* WtD3 = (u16*)take((size_t)1024 * 1024 * 2);
  u16* WtD4 = (u16*)take((size_t)1024 * 1024 * 2);
  u16* WtMu  = (u16*)take((size_t)32 * 1024 * 2);   // N=32, K=1024
  u16* WtLv  = (u16*)take((size_t)32 * 1024 * 2);
  u16* WtD1  = (u16*)take((size_t)1024 * 32 * 2);   // N=1024, K=32
  u16* WtB1  = (u16*)take((size_t)1024 * 32 * 2);   // ebm1: N=1024, K=32
  u16* WtB2  = (u16*)take((size_t)32 * 1024 * 2);   // ebm2: Npad=32, K=1024
  u16* WtDh  = (u16*)take((size_t)1024 * 32 * 2);   // ebm2 padded: N=1024,Kpad=32
  u16* WtB1T = (u16*)take((size_t)32 * 1024 * 2);   // ebm1 as-is: N=32, K=1024
  float* muF   = (float*)take((size_t)B_N * ZD_N * 4);
  float* lvF   = (float*)take((size_t)B_N * ZD_N * 4);
  float* zpr   = (float*)take((size_t)B_N * ZD_N * 4);
  float* dzF   = (float*)take((size_t)B_N * ZD_N * 4);
  u16*   zb    = (u16*)  take((size_t)B_N * ZD_N * 2);
  u16*   dlog  = (u16*)  take((size_t)B_N * 32 * 2);
  float* logit = (float*)take((size_t)B_N * NC_N * 4);
  float* acc   = (float*)take((size_t)ACC_COUNT * 4);

  const int tpb = 256;
  auto gridFor = [&](long n) { return dim3((unsigned)((n + tpb - 1) / tpb)); };

  // ---- init + weight prep (bf16, transposed to N-major) ----
  k_zero<<<1, ACC_COUNT, 0, stream>>>(acc, ACC_COUNT);
  k_cvt_bf16<<<gridFor((long)B_N * 1024), tpb, 0, stream>>>(x, bufA, (long)B_N * 1024);
  k_transpose_bf16<<<gridFor(1024L * 1024), tpb, 0, stream>>>(enc1_w, WtE1, 1024, 1024, 1024, 1024);
  k_transpose_bf16<<<gridFor(1024L * 1024), tpb, 0, stream>>>(enc2_w, WtE2, 1024, 1024, 1024, 1024);
  k_transpose_bf16<<<gridFor(1024L * 1024), tpb, 0, stream>>>(enc3_w, WtE3, 1024, 1024, 1024, 1024);
  k_transpose_bf16<<<gridFor(1024L * 1024), tpb, 0, stream>>>(dec2_w, WtD2, 1024, 1024, 1024, 1024);
  k_transpose_bf16<<<gridFor(1024L * 1024), tpb, 0, stream>>>(dec3_w, WtD3, 1024, 1024, 1024, 1024);
  k_transpose_bf16<<<gridFor(1024L * 1024), tpb, 0, stream>>>(dec4_w, WtD4, 1024, 1024, 1024, 1024);
  k_transpose_bf16<<<gridFor(32L * 1024), tpb, 0, stream>>>(mu_w,   WtMu, 1024, 32, 1024, 32);
  k_transpose_bf16<<<gridFor(32L * 1024), tpb, 0, stream>>>(lv_w,   WtLv, 1024, 32, 1024, 32);
  k_transpose_bf16<<<gridFor(1024L * 32), tpb, 0, stream>>>(dec1_w, WtD1, 32, 1024, 32, 1024);
  k_transpose_bf16<<<gridFor(1024L * 32), tpb, 0, stream>>>(ebm1_w, WtB1, 32, 1024, 32, 1024);
  k_transpose_bf16<<<gridFor(32L * 1024), tpb, 0, stream>>>(ebm2_w, WtB2, 1024, 20, 1024, 32);
  k_padcopy_bf16<<<gridFor(1024L * 32), tpb, 0, stream>>>(ebm2_w, WtDh, 20, 32, 1024);
  k_cvt_bf16<<<gridFor(32L * 1024), tpb, 0, stream>>>(ebm1_w, WtB1T, 32L * 1024);

  // ---- encoder ----
  run_lds<EPI_RELU_BF16>(bufA, WtE1, enc1_b, bufB, nullptr, nullptr, nullptr,
                         B_N, 1024, 1024, 1024, stream);
  run_lds<EPI_RELU_BF16>(bufB, WtE2, enc2_b, bufC, nullptr, nullptr, nullptr,
                         B_N, 1024, 1024, 1024, stream);
  run_lds<EPI_LIN_BF16>(bufC, WtE3, enc3_b, bufA, nullptr, nullptr, nullptr,
                        B_N, 1024, 1024, 1024, stream);               // h3 -> bufA
  run_direct<EPI_LIN_F32>(bufA, WtMu, mu_b, muF, B_N, 32, 1024, 32, 32, stream);
  run_direct<EPI_LIN_F32>(bufA, WtLv, lv_b, lvF, B_N, 32, 1024, 32, 32, stream);

  // ---- reparameterize + KL ----
  k_reparam_kl<<<gridFor((long)B_N * ZD_N), tpb, 0, stream>>>(muF, lvF, zb, acc);

  // ---- decoder + fused recon loss ----
  run_lds<EPI_RELU_BF16>(zb, WtD1, dec1_b, bufB, nullptr, nullptr, nullptr,
                         B_N, 1024, 32, 1024, stream);
  run_lds<EPI_RELU_BF16>(bufB, WtD2, dec2_b, bufC, nullptr, nullptr, nullptr,
                         B_N, 1024, 1024, 1024, stream);
  run_lds<EPI_RELU_BF16>(bufC, WtD3, dec3_b, bufB, nullptr, nullptr, nullptr,
                         B_N, 1024, 1024, 1024, stream);
  run_lds<EPI_SQDIFF>(bufB, WtD4, dec4_b, nullptr, nullptr, x, &acc[ACC_SQ],
                      B_N, 1024, 1024, 1024, stream);

  // ---- EBM on z_post: prob_pos + mutual information ----
  run_lds<EPI_GELU_PREACT>(zb, WtB1, ebm1_b, bufB /*act*/, bufA /*pre*/, nullptr,
                           nullptr, B_N, 1024, 32, 1024, stream);
  run_direct<EPI_LIN_F32>(bufB, WtB2, ebm2_b, logit, B_N, 32, 1024, NC_N, NC_N, stream);
  k_lse<<<B_N / tpb, tpb, 0, stream>>>(logit, &acc[ACC_POS], nullptr, acc);

  // ---- Langevin MCMC prior chain ----
  k_init_z<<<gridFor((long)B_N * ZD_N), tpb, 0, stream>>>(zpr, zb);
  for (int s = 0; s < NSTEPS; s++) {
    run_lds<EPI_GELU_PREACT>(zb, WtB1, ebm1_b, bufB, bufA, nullptr, nullptr,
                             B_N, 1024, 32, 1024, stream);
    run_direct<EPI_LIN_F32>(bufB, WtB2, ebm2_b, logit, B_N, 32, 1024, NC_N, NC_N, stream);
    k_lse<<<B_N / tpb, tpb, 0, stream>>>(logit, nullptr, dlog, nullptr);
    // dh = dlogits @ ebm2^T, times gelu'(pre)
    run_lds<EPI_GELUGRAD>(dlog, WtDh, nullptr, bufC, nullptr, bufA, nullptr,
                          B_N, 1024, 32, 1024, stream);
    // dz = dh @ ebm1^T
    run_direct<EPI_LIN_F32>(bufC, WtB1T, nullptr, dzF, B_N, 32, 1024, 32, 32, stream);
    k_langevin<<<gridFor((long)B_N * ZD_N), tpb, 0, stream>>>(zpr, dzF, zb, 100u + (u32)s);
  }

  // ---- prob_neg on z_prior ----
  run_lds<EPI_GELU_PREACT>(zb, WtB1, ebm1_b, bufB, bufA, nullptr, nullptr,
                           B_N, 1024, 32, 1024, stream);
  run_direct<EPI_LIN_F32>(bufB, WtB2, ebm2_b, logit, B_N, 32, 1024, NC_N, NC_N, stream);
  k_lse<<<B_N / tpb, tpb, 0, stream>>>(logit, &acc[ACC_NEG], nullptr, nullptr);

  // ---- combine into total loss ----
  k_finalize<<<1, 1, 0, stream>>>(acc, (float*)d_out);
}
